// MultiHeadAttention_52810917872237
// MI455X (gfx1250) — matrix operand order repaired and checked by reading, not compiled, not run on hardware
//
#include <hip/hip_runtime.h>
#include <hip/hip_bf16.h>

typedef __attribute__((ext_vector_type(16))) _Float16 v16h;
typedef __attribute__((ext_vector_type(8)))  _Float16 v8h;
typedef __attribute__((ext_vector_type(8)))  float    v8f;

#define BB 2
#define NN 2048
#define FF 1024
#define HH 16
#define DD 64
#define HD 1024

// ---------- WMMA fragment helpers (wave32, 16x16x32 f16) ----------

// A fragment: 16x32 (MxK) row-major f16, row stride ld (halfs).
// lane&15 = row, lane>>4 selects K-halves {0..15} vs {16..31}.
__device__ __forceinline__ v16h load_a_frag(const _Float16* base, int ld, int lane, int koff) {
    int r = lane & 15;
    int ksel = lane >> 4;
    const _Float16* p0 = base + (size_t)r * ld + koff + ksel * 16;
    const _Float16* p1 = base + (size_t)r * ld + koff + 8 + ksel * 16;
    v8h lo = *(const v8h*)p0;
    v8h hi = *(const v8h*)p1;
    v16h a;
#pragma unroll
    for (int i = 0; i < 8; ++i) { a[i] = lo[i]; a[8 + i] = hi[i]; }
    return a;
}

// B fragment: 32x16 (KxN). Operand stored K-contiguous per column:
// mat[col][k] with row stride ldk -> one contiguous 32B run per lane.
__device__ __forceinline__ v16h load_b_frag(const _Float16* matT, int ldk, int nbase, int lane, int koff) {
    int n = nbase + (lane & 15);
    int ksel = lane >> 4;
    return *(const v16h*)(matT + (size_t)n * ldk + koff + ksel * 16);
}

#define WMMA_F16(Afrag, Bfrag, Cacc) \
    __builtin_amdgcn_wmma_f32_16x16x32_f16(false, (Afrag), false, (Bfrag), (short)0, (Cacc), false, false)

// ---------- conversion kernels ----------

__global__ __launch_bounds__(256) void cvt_f32_to_f16(const float* __restrict__ src,
                                                      _Float16* __restrict__ dst, int n) {
    int i = blockIdx.x * 256 + threadIdx.x;
    if (i < n) dst[i] = (_Float16)src[i];
}

// src [rows, cols] f32 -> dst [cols, rows] f16 (transpose)
__global__ __launch_bounds__(256) void cvt_transpose_f16(const float* __restrict__ src,
                                                         _Float16* __restrict__ dst,
                                                         int rows, int cols) {
    int i = blockIdx.x * 256 + threadIdx.x;
    if (i < rows * cols) {
        int r = i / cols, c = i - r * cols;
        dst[(size_t)c * rows + r] = (_Float16)src[i];
    }
}

// ---------- QKV projection GEMM: Y = x @ W + b, heads-major f16 outputs ----------
// blockIdx.y: 0 = Q, 1 = K (store [B,H,N,D]), 2 = V (store transposed [B,H,D,N])
// Each wave: 16 rows x 64 cols, K double-buffered.

__global__ __launch_bounds__(256) void qkv_gemm_kernel(
    const _Float16* __restrict__ xh,
    const _Float16* __restrict__ wqt, const _Float16* __restrict__ wkt, const _Float16* __restrict__ wvt,
    const float* __restrict__ bq, const float* __restrict__ bk, const float* __restrict__ bv,
    _Float16* __restrict__ qo, _Float16* __restrict__ ko, _Float16* __restrict__ vo)
{
    int which = blockIdx.y;
    const _Float16* wt = (which == 0) ? wqt : (which == 1) ? wkt : wvt;
    const float* bias  = (which == 0) ? bq  : (which == 1) ? bk  : bv;
    _Float16* out      = (which == 0) ? qo  : (which == 1) ? ko  : vo;

    int lane = threadIdx.x & 31;
    int wid  = (blockIdx.x * 256 + threadIdx.x) >> 5;  // 0..4095
    int mt = wid >> 4;        // 0..255 : 16-row tile
    int cs = wid & 15;        // 0..15  : 64-col strip
    int nb = cs * 64;

    const _Float16* abase = xh + (size_t)mt * 16 * FF;
    v8f acc[4] = {{}, {}, {}, {}};

    // double-buffered K loop
    v16h a_c  = load_a_frag(abase, FF, lane, 0);
    v16h b_c0 = load_b_frag(wt, FF, nb,      lane, 0);
    v16h b_c1 = load_b_frag(wt, FF, nb + 16, lane, 0);
    v16h b_c2 = load_b_frag(wt, FF, nb + 32, lane, 0);
    v16h b_c3 = load_b_frag(wt, FF, nb + 48, lane, 0);
    for (int kk = 0; kk < FF - 32; kk += 32) {
        int kn = kk + 32;
        v16h a_n  = load_a_frag(abase, FF, lane, kn);
        v16h b_n0 = load_b_frag(wt, FF, nb,      lane, kn);
        v16h b_n1 = load_b_frag(wt, FF, nb + 16, lane, kn);
        v16h b_n2 = load_b_frag(wt, FF, nb + 32, lane, kn);
        v16h b_n3 = load_b_frag(wt, FF, nb + 48, lane, kn);
        acc[0] = WMMA_F16(a_c, b_c0, acc[0]);
        acc[1] = WMMA_F16(a_c, b_c1, acc[1]);
        acc[2] = WMMA_F16(a_c, b_c2, acc[2]);
        acc[3] = WMMA_F16(a_c, b_c3, acc[3]);
        a_c = a_n; b_c0 = b_n0; b_c1 = b_n1; b_c2 = b_n2; b_c3 = b_n3;
    }
    acc[0] = WMMA_F16(a_c, b_c0, acc[0]);
    acc[1] = WMMA_F16(a_c, b_c1, acc[1]);
    acc[2] = WMMA_F16(a_c, b_c2, acc[2]);
    acc[3] = WMMA_F16(a_c, b_c3, acc[3]);

    int mbase = (lane >> 4) * 8;
    int ncl = lane & 15;
#pragma unroll
    for (int t = 0; t < 4; ++t) {
#pragma unroll
        for (int j = 0; j < 8; ++j) {
            int row = mt * 16 + mbase + j;            // 0..4095 = b*N + n
            int col = nb + t * 16 + ncl;              // 0..1023 = h*64 + d
            float v = acc[t][j] + bias[col];
            int b = row >> 11, n = row & (NN - 1);
            int h = col >> 6,  d = col & (DD - 1);
            size_t idx;
            if (which == 2) idx = (((size_t)(b * HH + h) * DD + d) * NN + n);     // V transposed
            else            idx = (((size_t)(b * HH + h) * NN + n) * DD + d);     // Q,K
            out[idx] = (_Float16)v;
        }
    }
}

// ---------- fused scores + softmax + attn@V ----------
// one block per (b, h, 16-row tile): grid = B*H*(N/16) = 4096, 256 threads (8 waves)

__global__ __launch_bounds__(256) void attn_kernel(
    const _Float16* __restrict__ qh, const _Float16* __restrict__ kh, const _Float16* __restrict__ vt,
    const int* __restrict__ mask, float* __restrict__ attn_out, _Float16* __restrict__ oh)
{
    __shared__ float    sc[16 * 2048];     // 128 KB raw scores / exp values
    __shared__ _Float16 ph[16 * 2048];     //  64 KB normalized probs (f16, A-frag source)
    __shared__ float red[16 * 16];
    __shared__ float rowmax[16];
    __shared__ float rowsum[16];
    __shared__ float oacc[16 * 64];

    int bid = blockIdx.x;
    int mt = bid & 127;
    int h  = (bid >> 7) & 15;
    int b  = bid >> 11;
    int tid = threadIdx.x;
    int lane = tid & 31;
    int wave = tid >> 5;

    const _Float16* qbase = qh + (((size_t)(b * HH + h) * NN) + mt * 16) * DD;
    const _Float16* kbase = kh + ((size_t)(b * HH + h) * NN) * DD;
    const _Float16* vbase = vt + ((size_t)(b * HH + h) * DD) * NN;

    // Q fragments for this 16-row tile, K-depth 64 split into two 32-steps
    v16h a0 = load_a_frag(qbase, DD, lane, 0);
    v16h a1 = load_a_frag(qbase, DD, lane, 32);

    int mbase = (lane >> 4) * 8;
    int ncl = lane & 15;

    // scores = Q @ K^T / 8, masked, into LDS (double-buffered over n-tiles)
    {
        int nt = wave;
        v16h kb0 = load_b_frag(kbase, DD, nt * 16, lane, 0);
        v16h kb1 = load_b_frag(kbase, DD, nt * 16, lane, 32);
        while (nt < NN / 16) {
            int ntn = nt + 8;
            int ntp = (ntn < NN / 16) ? ntn : nt;   // clamp: harmless reload on last iter
            v16h nb0 = load_b_frag(kbase, DD, ntp * 16, lane, 0);
            v16h nb1 = load_b_frag(kbase, DD, ntp * 16, lane, 32);
            v8f c = {};
            c = WMMA_F16(a0, kb0, c);
            c = WMMA_F16(a1, kb1, c);
#pragma unroll
            for (int j = 0; j < 8; ++j) {
                int m = mbase + j;
                int col = nt * 16 + ncl;
                float s = c[j] * 0.125f;   // 1/sqrt(64)
                int mk = mask[((size_t)b * NN + mt * 16 + m) * NN + col];
                if (mk == 0) s = -9.0e15f;
                sc[m * 2048 + col] = s;
            }
            kb0 = nb0; kb1 = nb1; nt = ntn;
        }
    }
    __syncthreads();

    // softmax over each of the 16 rows (16 threads per row)
    int row = tid >> 4;
    int sub = tid & 15;
    float mx = -3.4e38f;
    for (int c0 = sub; c0 < NN; c0 += 16) mx = fmaxf(mx, sc[row * 2048 + c0]);
    red[row * 16 + sub] = mx;
    __syncthreads();
    if (sub == 0) {
        float m2 = red[row * 16];
        for (int i = 1; i < 16; ++i) m2 = fmaxf(m2, red[row * 16 + i]);
        rowmax[row] = m2;
    }
    __syncthreads();
    float rm = rowmax[row];
    float sm = 0.f;
    for (int c0 = sub; c0 < NN; c0 += 16) {
        float e = __expf(sc[row * 2048 + c0] - rm);
        sc[row * 2048 + c0] = e;
        sm += e;
    }
    red[row * 16 + sub] = sm;
    __syncthreads();
    if (sub == 0) {
        float s2 = 0.f;
        for (int i = 0; i < 16; ++i) s2 += red[row * 16 + i];
        rowsum[row] = s2;
    }
    __syncthreads();

    // normalize: write attn (f32) to d_out, stash f16 probs in LDS for attn@V
    float inv = 1.0f / rowsum[row];
    float* arow = attn_out + (((size_t)(b * HH + h) * NN) + mt * 16 + row) * NN;
    for (int c0 = sub; c0 < NN; c0 += 16) {
        float p = sc[row * 2048 + c0] * inv;
        ph[row * 2048 + c0] = (_Float16)p;
        arow[c0] = p;
    }
    for (int i = tid; i < 16 * 64; i += 256) oacc[i] = 0.f;
    __syncthreads();

    // out_tile[16x64] = attn_tile[16x2048] @ V[2048x64]
    // 8 waves: column tile = wave&3, K-half = wave>>2; combine via LDS atomics
    {
        int ct = wave & 3;
        int khalf = wave >> 2;
        int kbeg = khalf * 1024, kend = kbeg + 1024;
        v8f c = {};
        v16h a_c = load_a_frag(ph, 2048, lane, kbeg);
        v16h b_c = load_b_frag(vbase, NN, ct * 16, lane, kbeg);
        for (int kk = kbeg; kk < kend; kk += 32) {
            int kn = (kk + 32 < kend) ? kk + 32 : kk;
            v16h a_n = load_a_frag(ph, 2048, lane, kn);
            v16h b_n = load_b_frag(vbase, NN, ct * 16, lane, kn);
            c = WMMA_F16(a_c, b_c, c);
            a_c = a_n; b_c = b_n;
        }
#pragma unroll
        for (int j = 0; j < 8; ++j) {
            int m = mbase + j;
            int d = ct * 16 + ncl;
            atomicAdd(&oacc[m * 64 + d], c[j]);
        }
    }
    __syncthreads();

    // store O tile as f16 back in [B*N, H*D] layout for the output projection
    for (int i = tid; i < 16 * 64; i += 256) {
        int m = i >> 6, d = i & 63;
        size_t orow = (size_t)b * NN + mt * 16 + m;
        oh[orow * HD + h * DD + d] = (_Float16)oacc[i];
    }
}

// ---------- output projection: out = O @ Wo + bo (f32 result) ----------

__global__ __launch_bounds__(256) void out_gemm_kernel(
    const _Float16* __restrict__ oh, const _Float16* __restrict__ wot,
    const float* __restrict__ bo, float* __restrict__ out)
{
    int lane = threadIdx.x & 31;
    int wid  = (blockIdx.x * 256 + threadIdx.x) >> 5;  // 0..4095
    int mt = wid >> 4;
    int cs = wid & 15;
    int nb = cs * 64;

    const _Float16* abase = oh + (size_t)mt * 16 * HD;
    v8f acc[4] = {{}, {}, {}, {}};

    v16h a_c  = load_a_frag(abase, HD, lane, 0);
    v16h b_c0 = load_b_frag(wot, HD, nb,      lane, 0);
    v16h b_c1 = load_b_frag(wot, HD, nb + 16, lane, 0);
    v16h b_c2 = load_b_frag(wot, HD, nb + 32, lane, 0);
    v16h b_c3 = load_b_frag(wot, HD, nb + 48, lane, 0);
    for (int kk = 0; kk < HD - 32; kk += 32) {
        int kn = kk + 32;
        v16h a_n  = load_a_frag(abase, HD, lane, kn);
        v16h b_n0 = load_b_frag(wot, HD, nb,      lane, kn);
        v16h b_n1 = load_b_frag(wot, HD, nb + 16, lane, kn);
        v16h b_n2 = load_b_frag(wot, HD, nb + 32, lane, kn);
        v16h b_n3 = load_b_frag(wot, HD, nb + 48, lane, kn);
        acc[0] = WMMA_F16(a_c, b_c0, acc[0]);
        acc[1] = WMMA_F16(a_c, b_c1, acc[1]);
        acc[2] = WMMA_F16(a_c, b_c2, acc[2]);
        acc[3] = WMMA_F16(a_c, b_c3, acc[3]);
        a_c = a_n; b_c0 = b_n0; b_c1 = b_n1; b_c2 = b_n2; b_c3 = b_n3;
    }
    acc[0] = WMMA_F16(a_c, b_c0, acc[0]);
    acc[1] = WMMA_F16(a_c, b_c1, acc[1]);
    acc[2] = WMMA_F16(a_c, b_c2, acc[2]);
    acc[3] = WMMA_F16(a_c, b_c3, acc[3]);

    int mbase = (lane >> 4) * 8;
    int ncl = lane & 15;
#pragma unroll
    for (int t = 0; t < 4; ++t) {
#pragma unroll
        for (int j = 0; j < 8; ++j) {
            int row = mt * 16 + mbase + j;
            int col = nb + t * 16 + ncl;
            out[(size_t)row * FF + col] = acc[t][j] + bo[col];
        }
    }
}

// ---------- host launch ----------

extern "C" void kernel_launch(void* const* d_in, const int* in_sizes, int n_in,
                              void* d_out, int out_size, void* d_ws, size_t ws_size,
                              hipStream_t stream) {
    const float* x    = (const float*)d_in[0];
    const int*   mask = (const int*)d_in[1];
    const float* Wq   = (const float*)d_in[2];
    const float* bq   = (const float*)d_in[3];
    const float* Wk   = (const float*)d_in[4];
    const float* bk   = (const float*)d_in[5];
    const float* Wv   = (const float*)d_in[6];
    const float* bv   = (const float*)d_in[7];
    const float* Wo   = (const float*)d_in[8];
    const float* bo   = (const float*)d_in[9];

    float* out  = (float*)d_out;
    float* attn = out + (size_t)BB * NN * HD;          // out first, attn second

    char* ws = (char*)d_ws;
    const size_t X_ELEMS = (size_t)BB * NN * FF;       // 4M
    const size_t W_ELEMS = (size_t)FF * HD;            // 1M
    _Float16* xh  = (_Float16*)ws;  ws += X_ELEMS * 2;
    _Float16* wqt = (_Float16*)ws;  ws += W_ELEMS * 2;
    _Float16* wkt = (_Float16*)ws;  ws += W_ELEMS * 2;
    _Float16* wvt = (_Float16*)ws;  ws += W_ELEMS * 2;
    _Float16* wot = (_Float16*)ws;  ws += W_ELEMS * 2;
    _Float16* qh  = (_Float16*)ws;  ws += X_ELEMS * 2;
    _Float16* kh  = (_Float16*)ws;  ws += X_ELEMS * 2;
    _Float16* vt  = (_Float16*)ws;  ws += X_ELEMS * 2;
    _Float16* oh  = (_Float16*)ws;  ws += X_ELEMS * 2;

    cvt_f32_to_f16<<<(int)(X_ELEMS / 256), 256, 0, stream>>>(x, xh, (int)X_ELEMS);
    cvt_transpose_f16<<<(int)(W_ELEMS / 256), 256, 0, stream>>>(Wq, wqt, FF, HD);
    cvt_transpose_f16<<<(int)(W_ELEMS / 256), 256, 0, stream>>>(Wk, wkt, FF, HD);
    cvt_transpose_f16<<<(int)(W_ELEMS / 256), 256, 0, stream>>>(Wv, wvt, FF, HD);
    cvt_transpose_f16<<<(int)(W_ELEMS / 256), 256, 0, stream>>>(Wo, wot, HD, FF);

    qkv_gemm_kernel<<<dim3(512, 3), 256, 0, stream>>>(xh, wqt, wkt, wvt, bq, bk, bv, qh, kh, vt);
    attn_kernel<<<BB * HH * (NN / 16), 256, 0, stream>>>(qh, kh, vt, mask, attn, oh);
    out_gemm_kernel<<<512, 256, 0, stream>>>(oh, wot, bo, out);
}
